// LinearKAN_67199058313833
// MI455X (gfx1250) — matrix operand order repaired and checked by reading, not compiled, not run on hardware
//
#include <hip/hip_runtime.h>

// ---- types for WMMA fragments (gfx1250, wave32) ----
typedef __bf16 v16bf  __attribute__((ext_vector_type(16)));
typedef __bf16 bf16x8 __attribute__((ext_vector_type(8)));
typedef __bf16 bf16x2 __attribute__((ext_vector_type(2)));
typedef float  v8f    __attribute__((ext_vector_type(8)));

union Frag { v16bf v; bf16x8 h[2]; };

#define N_IN          64
#define N_OUT         64
#define NBASIS        8
#define KTOT          576      // 9 features * 64 inputs
#define KCHUNKS       18       // 576 / 32
#define ASTRIDE       584      // padded LDS row stride (bf16 elements), 16B aligned
#define ROWS_PER_WAVE 32
#define WAVES         4
#define ROWS_PER_BLOCK 128

// ---- prep: fold weights into bf16 W^T[n][f*64+i] in scratch ----
__global__ void kan_prep_weights(const float* __restrict__ wb,
                                 const float* __restrict__ wsp,
                                 const float* __restrict__ c,
                                 __bf16* __restrict__ bt) {
  int idx = blockIdx.x * blockDim.x + threadIdx.x;
  if (idx >= N_OUT * KTOT) return;
  int n  = idx / KTOT;
  int kk = idx - n * KTOT;
  int f  = kk >> 6;          // 0 = silu channel, 1..8 = Hermite H0..H7
  int i  = kk & 63;
  float v;
  if (f == 0) v = wb[i * N_OUT + n];
  else        v = wsp[i * N_OUT + n] * c[(i * N_OUT + n) * NBASIS + (f - 1)];
  bt[idx] = (__bf16)v;
}

// silu + physicists' Hermite H0..H7
__device__ __forceinline__ void kan_features(float xx, float f[9]) {
  f[0] = xx / (1.0f + __expf(-xx));
  float hm2 = 1.0f;
  float hm1 = 2.0f * xx;
  f[1] = hm2;
  f[2] = hm1;
#pragma unroll
  for (int k = 2; k < NBASIS; ++k) {
    float h = 2.0f * xx * hm1 - 2.0f * (float)(k - 1) * hm2;
    hm2 = hm1; hm1 = h;
    f[k + 1] = h;
  }
}

__global__ __launch_bounds__(128)
void kan_main(const float* __restrict__ x,
              const __bf16* __restrict__ bt,
              float* __restrict__ out) {
  extern __shared__ __bf16 ldsA[];          // WAVES * 32 * ASTRIDE bf16
  const int lane = threadIdx.x & 31;
  const int wave = threadIdx.x >> 5;
  const int b0   = blockIdx.x * ROWS_PER_BLOCK + wave * ROWS_PER_WAVE;
  __bf16* A = ldsA + wave * ROWS_PER_WAVE * ASTRIDE;

  // ---- phase 1: build 32 x 576 bf16 activation tile in LDS ----
#pragma unroll 2
  for (int m = 0; m < ROWS_PER_WAVE; ++m) {
    const float2 xv = *(const float2*)(x + (size_t)(b0 + m) * N_IN + 2 * lane);
    if (m + 1 < ROWS_PER_WAVE)
      __builtin_prefetch(x + (size_t)(b0 + m + 1) * N_IN + 2 * lane, 0, 0);
    float f0[9], f1[9];
    kan_features(xv.x, f0);
    kan_features(xv.y, f1);
    __bf16* rowp = A + m * ASTRIDE + 2 * lane;
#pragma unroll
    for (int f = 0; f < 9; ++f) {
      bf16x2 p;
      p.x = (__bf16)f0[f];
      p.y = (__bf16)f1[f];
      *(bf16x2*)(rowp + f * 64) = p;          // ds_store_b32, conflict-free
    }
  }
  __syncthreads();

  // ---- phase 2: (32 x 576) x (576 x 64) via v_wmma_f32_16x16x32_bf16 ----
  v8f acc0[4], acc1[4];
  const v8f vzero = {0.f, 0.f, 0.f, 0.f, 0.f, 0.f, 0.f, 0.f};
#pragma unroll
  for (int nt = 0; nt < 4; ++nt) { acc0[nt] = vzero; acc1[nt] = vzero; }

  const int mrow  = lane & 15;   // M index within tile
  const int khalf = lane >> 4;   // which K-half this lane holds

  for (int ck = 0; ck < KCHUNKS; ++ck) {
    const int kbA = ck * 32 + khalf * 8;     // A: lanes 0-15 hold K 0..7/16..23, lanes 16-31 K 8..15/24..31
    Frag a0, a1;
    a0.h[0] = *(const bf16x8*)(A + mrow * ASTRIDE + kbA);
    a0.h[1] = *(const bf16x8*)(A + mrow * ASTRIDE + kbA + 16);
    a1.h[0] = *(const bf16x8*)(A + (16 + mrow) * ASTRIDE + kbA);
    a1.h[1] = *(const bf16x8*)(A + (16 + mrow) * ASTRIDE + kbA + 16);

    const int kbB = ck * 32 + khalf * 8;     // B: lanes 0-15 K 0..7/16..23, lanes 16-31 K 8..15/24..31 (col N = lane%16)
#pragma unroll
    for (int nt = 0; nt < 4; ++nt) {
      Frag bfr;
      const __bf16* bp = bt + (size_t)(nt * 16 + mrow) * KTOT + kbB;
      bfr.h[0] = *(const bf16x8*)(bp);
      bfr.h[1] = *(const bf16x8*)(bp + 16);
      acc0[nt] = __builtin_amdgcn_wmma_f32_16x16x32_bf16(
          false, a0.v, false, bfr.v, (short)0, acc0[nt], false, false);
      acc1[nt] = __builtin_amdgcn_wmma_f32_16x16x32_bf16(
          false, a1.v, false, bfr.v, (short)0, acc1[nt], false, false);
    }
  }

  // ---- phase 3: store f32 results (C/D layout: VGPR v -> M = v + 8*(lane/16), N = lane%16) ----
  const int col  = lane & 15;
  const int rofs = (lane >> 4) * 8;
#pragma unroll
  for (int nt = 0; nt < 4; ++nt) {
#pragma unroll
    for (int v = 0; v < 8; ++v) {
      out[(size_t)(b0 + rofs + v) * N_OUT + nt * 16 + col]       = acc0[nt][v];
      out[(size_t)(b0 + 16 + rofs + v) * N_OUT + nt * 16 + col]  = acc1[nt][v];
    }
  }
}

extern "C" void kernel_launch(void* const* d_in, const int* in_sizes, int n_in,
                              void* d_out, int out_size, void* d_ws, size_t ws_size,
                              hipStream_t stream) {
  const float* x   = (const float*)d_in[0];
  const float* wb  = (const float*)d_in[1];
  const float* wsp = (const float*)d_in[2];
  const float* c   = (const float*)d_in[3];
  float*  out = (float*)d_out;
  __bf16* bt  = (__bf16*)d_ws;   // 64*576*2 = 73728 bytes of scratch

  const int prep_elems = N_OUT * KTOT;
  kan_prep_weights<<<(prep_elems + 255) / 256, 256, 0, stream>>>(wb, wsp, c, bt);

  const int B    = in_sizes[0] / N_IN;        // 131072
  const int grid = B / ROWS_PER_BLOCK;        // 1024
  const size_t lds_bytes = (size_t)WAVES * ROWS_PER_WAVE * ASTRIDE * sizeof(__bf16); // 149504
  kan_main<<<grid, 128, lds_bytes, stream>>>(x, bt, out);
}
